// MultiHeadAttention_14482629722864
// MI455X (gfx1250) — compile-verified, matrix operand order checked by reading, not run on hardware
//
#include <hip/hip_runtime.h>

// MI455X (gfx1250, wave32) fused MHA forward.
// ~137 GFLOP vs ~80MB minimal HBM traffic @23.3TB/s -> compute-bound, so all
// matmuls use v_wmma_f32_16x16x32_f16 (f16 in, f32 accumulate). This version
// adds software pipelining (so wmma overlaps in-flight loads instead of
// s_wait_loadcnt 0 before every MMA), LDS staging of shared activation
// panels, and the CDNA5 async global->LDS copy path (ASYNCcnt).

typedef _Float16 h16;
typedef __attribute__((ext_vector_type(4)))  _Float16 v4h;
typedef __attribute__((ext_vector_type(8)))  _Float16 v8h;
typedef __attribute__((ext_vector_type(16))) _Float16 v16h;
typedef __attribute__((ext_vector_type(8)))  float    v8f;
typedef __attribute__((ext_vector_type(4)))  int      i4;

#define B_ 4
#define N_ 2048
#define E_ 1024
#define H_ 16
#define D_ 64
#define BN_ (B_*N_)
#define APAD 1032   // 1024 + 8 halfs row padding: row stride 2064B -> distinct LDS bank groups

#if __has_builtin(__builtin_amdgcn_global_load_async_to_lds_b128)
#define HAVE_ASYNC_LDS 1
#else
#define HAVE_ASYNC_LDS 0
#endif

typedef __attribute__((address_space(1))) i4* gv4i;   // global int32x4*
typedef __attribute__((address_space(3))) i4* lv4i;   // LDS int32x4*

// 16-byte global->LDS copy: async (ASYNCcnt-tracked) when available.
__device__ __forceinline__ void cp16_g2l(h16* lds_dst, const h16* gsrc) {
#if HAVE_ASYNC_LDS
  __builtin_amdgcn_global_load_async_to_lds_b128(
      (gv4i)(uintptr_t)gsrc,
      (lv4i)(unsigned)(uintptr_t)lds_dst,   // low 32 bits = LDS offset
      0, 0);
#else
  *reinterpret_cast<v8h*>(lds_dst) = *reinterpret_cast<const v8h*>(gsrc);
#endif
}

__device__ __forceinline__ void wait_async0() {
#if __has_builtin(__builtin_amdgcn_s_wait_asynccnt)
  __builtin_amdgcn_s_wait_asynccnt(0);
#else
  asm volatile("s_wait_asynccnt 0x0" ::: "memory");
#endif
}

// Load one 16x32 f16 operand tile (A-row or B-column striped) in the CDNA5
// WMMA register layout: lane<16 holds K {k0..k0+7, k0+16..k0+23},
// lane>=16 holds K {k0+8..k0+15, k0+24..k0+31}; rc = absolute row/column.
__device__ __forceinline__ v16h ld_tile(const h16* __restrict__ p, int ld, int rc, int k0, int hf) {
  const v8h* p0 = reinterpret_cast<const v8h*>(p + (size_t)rc * ld + k0 + hf * 8);
  const v8h* p1 = reinterpret_cast<const v8h*>(p + (size_t)rc * ld + k0 + 16 + hf * 8);
  v8h lo = *p0, hi = *p1;
  return __builtin_shufflevector(lo, hi, 0,1,2,3,4,5,6,7,8,9,10,11,12,13,14,15);
}

// Same tile shape but sourced from an LDS row (ds_load_b128 x2).
__device__ __forceinline__ v16h lds_a(const h16* arow, int k0, int hf) {
  const v8h* p0 = reinterpret_cast<const v8h*>(arow + k0 + hf * 8);
  const v8h* p1 = reinterpret_cast<const v8h*>(arow + k0 + 16 + hf * 8);
  v8h lo = *p0, hi = *p1;
  return __builtin_shufflevector(lo, hi, 0,1,2,3,4,5,6,7,8,9,10,11,12,13,14,15);
}

__device__ __forceinline__ v8f wmma16(v16h a, v16h b, v8f c) {
  return __builtin_amdgcn_wmma_f32_16x16x32_f16(false, a, false, b, (short)0, c, false, false);
}

// ---------------------------------------------------------------- convert ---
__global__ void k_f32_to_f16(const float* __restrict__ in, h16* __restrict__ out, int n) {
  int i = (blockIdx.x * blockDim.x + threadIdx.x) * 4;
  if (i + 3 < n) {
    float4 v = *reinterpret_cast<const float4*>(in + i);
    v4h o = { (h16)v.x, (h16)v.y, (h16)v.z, (h16)v.w };
    *reinterpret_cast<v4h*>(out + i) = o;
  } else {
    for (; i < n; ++i) out[i] = (h16)in[i];
  }
}

// Cooperative stage of a 16x1024 f16 activation panel into padded LDS.
__device__ __forceinline__ void stage_panel(h16 (*As)[APAD], const h16* src16rows, int tid) {
  // 16 rows x 1024 halfs = 2048 16-byte chunks; 256 threads x 8 chunks.
  for (int i = 0; i < 8; ++i) {
    int c = tid + 256 * i;
    int row = c >> 7;            // 128 chunks per row
    int col = (c & 127) * 8;
    cp16_g2l(&As[row][col], src16rows + (size_t)row * E_ + col);
  }
#if HAVE_ASYNC_LDS
  wait_async0();
#endif
  __syncthreads();
}

// ------------------------------------------------- QKV projection + RoPE ---
// grid (512, 6) x 256 threads. All 8 waves of a block share one 16-row
// activation panel staged in LDS. Each wave computes 16 rows x 64 features
// of Q, K or V; K-loop is software-pipelined (prefetch k+32 during wmma).
__global__ void __launch_bounds__(256) k_qkv_rope(
    const h16* __restrict__ xh, const h16* __restrict__ Wq,
    const h16* __restrict__ Wk, const h16* __restrict__ Wv,
    h16* __restrict__ Qh, h16* __restrict__ Kh, h16* __restrict__ Vt)
{
  __shared__ h16 As[16][APAD];
  const int tid  = threadIdx.x;
  const int lane = tid & 31;
  const int wv   = tid >> 5;
  const int hf   = lane >> 4;
  const int lm   = lane & 15;
  const int rt   = blockIdx.x;              // 0..511 row tiles of BN
  const int idx  = blockIdx.y * 8 + wv;     // 0..47
  const int mat  = idx >> 4;                // 0=Q 1=K 2=V
  const int ft   = idx & 15;                // head index (64 features each)
  const int f0   = ft * 64;
  const h16* W = (mat == 0) ? Wq : (mat == 1) ? Wk : Wv;

  stage_panel(As, xh + (size_t)rt * 16 * E_, tid);
  const h16* arow = &As[lm][0];

  v8f c0 = {}, c1 = {}, c2 = {}, c3 = {};
  v16h a_c  = lds_a(arow, 0, hf);
  v16h b0_c = ld_tile(W, E_, f0 +  0 + lm, 0, hf);
  v16h b1_c = ld_tile(W, E_, f0 + 16 + lm, 0, hf);
  v16h b2_c = ld_tile(W, E_, f0 + 32 + lm, 0, hf);
  v16h b3_c = ld_tile(W, E_, f0 + 48 + lm, 0, hf);
  for (int k0 = 0; k0 < E_ - 32; k0 += 32) {
    const int kn = k0 + 32;
    v16h a_n  = lds_a(arow, kn, hf);
    v16h b0_n = ld_tile(W, E_, f0 +  0 + lm, kn, hf);
    v16h b1_n = ld_tile(W, E_, f0 + 16 + lm, kn, hf);
    v16h b2_n = ld_tile(W, E_, f0 + 32 + lm, kn, hf);
    v16h b3_n = ld_tile(W, E_, f0 + 48 + lm, kn, hf);
    c0 = wmma16(a_c, b0_c, c0);
    c1 = wmma16(a_c, b1_c, c1);
    c2 = wmma16(a_c, b2_c, c2);
    c3 = wmma16(a_c, b3_c, c3);
    a_c = a_n; b0_c = b0_n; b1_c = b1_n; b2_c = b2_n; b3_c = b3_n;
  }
  c0 = wmma16(a_c, b0_c, c0);
  c1 = wmma16(a_c, b1_c, c1);
  c2 = wmma16(a_c, b2_c, c2);
  c3 = wmma16(a_c, b3_c, c3);

  const int h = ft;
  if (mat == 2) {
    // V transposed: Vt[((b*H+h)*D + d)*N + n]
    for (int v = 0; v < 8; ++v) {
      int grow = rt * 16 + v + 8 * hf;
      int b = grow >> 11, n = grow & (N_ - 1);
      size_t base = (((size_t)b * H_ + h) * D_) * N_ + n;
      Vt[base + (size_t)( 0 + lm) * N_] = (h16)c0[v];
      Vt[base + (size_t)(16 + lm) * N_] = (h16)c1[v];
      Vt[base + (size_t)(32 + lm) * N_] = (h16)c2[v];
      Vt[base + (size_t)(48 + lm) * N_] = (h16)c3[v];
    }
  } else {
    h16* dst = (mat == 0) ? Qh : Kh;
    // inv_freq = 10000^(-j/32), j = d % 32; ln(10000)/32 = 0.28782314
    const float if0 = __expf(-0.28782314f * (float)lm);
    const float if1 = __expf(-0.28782314f * (float)(lm + 16));
    for (int v = 0; v < 8; ++v) {
      int grow = rt * 16 + v + 8 * hf;
      int b = grow >> 11, n = grow & (N_ - 1);
      float s0, cs0, s1, cs1;
      __sincosf((float)n * if0, &s0, &cs0);
      __sincosf((float)n * if1, &s1, &cs1);
      float q0 = c0[v] * cs0 - c2[v] * s0;   // d = lm      (<32): -pair(d+32)
      float q1 = c1[v] * cs1 - c3[v] * s1;   // d = 16+lm   (<32)
      float q2 = c2[v] * cs0 + c0[v] * s0;   // d = 32+lm   (>=32): +pair(d-32)
      float q3 = c3[v] * cs1 + c1[v] * s1;   // d = 48+lm
      size_t base = ((((size_t)b * H_ + h) * N_) + n) * D_;
      dst[base +  0 + lm] = (h16)q0;
      dst[base + 16 + lm] = (h16)q1;
      dst[base + 32 + lm] = (h16)q2;
      dst[base + 48 + lm] = (h16)q3;
    }
  }
}

// ----------------------------------------------------- flash attention -----
// One wave per 16 query rows of one (b,h). Online softmax. V-loads and the
// next iteration's K-loads are issued before the softmax VALU/exp work so
// transcendental latency hides memory latency; P goes through a 1KB
// per-wave LDS bounce to convert C-layout -> A-layout for the PV WMMA.
__global__ void __launch_bounds__(256) k_attn(
    const h16* __restrict__ Qh, const h16* __restrict__ Kh,
    const h16* __restrict__ Vt, h16* __restrict__ CTX)
{
  __shared__ h16 lds[8][16 * 32];
  const int lane = threadIdx.x & 31;
  const int wv   = threadIdx.x >> 5;
  const int hf   = lane >> 4, lm = lane & 15;
  const int w  = blockIdx.x * 8 + wv;       // 0..8191
  const int bh = w >> 7;
  const int qt = w & 127;
  const int b = bh >> 4, h = bh & 15;
  const h16* Qbh = Qh + (((size_t)b * H_ + h) * N_) * D_;
  const h16* Kbh = Kh + (((size_t)b * H_ + h) * N_) * D_;
  const h16* Vbh = Vt + (((size_t)b * H_ + h) * D_) * N_;
  const int n0 = qt * 16;

  v16h aq0 = ld_tile(Qbh, D_, n0 + lm,  0, hf);
  v16h aq1 = ld_tile(Qbh, D_, n0 + lm, 32, hf);

  v8f o0 = {}, o1 = {}, o2 = {}, o3 = {};
  float mrow[8], lrow[8];
  for (int v = 0; v < 8; ++v) { mrow[v] = -3.0e38f; lrow[v] = 0.0f; }
  const float scale = 0.125f;  // 1/sqrt(64)

  // Preload K tiles for j0 = 0.
  v16h k00 = ld_tile(Kbh, D_,      lm,  0, hf);
  v16h k01 = ld_tile(Kbh, D_,      lm, 32, hf);
  v16h k10 = ld_tile(Kbh, D_, 16 + lm,  0, hf);
  v16h k11 = ld_tile(Kbh, D_, 16 + lm, 32, hf);

  for (int j0 = 0; j0 < N_; j0 += 32) {
    // S = Q K^T on the prefetched tiles.
    v8f s0 = {}, s1 = {};
    s0 = wmma16(aq0, k00, s0); s0 = wmma16(aq1, k01, s0);
    s1 = wmma16(aq0, k10, s1); s1 = wmma16(aq1, k11, s1);

    // Issue V loads for this step and K loads for the next step NOW so they
    // overlap the softmax VALU below.
    v16h bv0 = ld_tile(Vbh, N_,  0 + lm, j0, hf);
    v16h bv1 = ld_tile(Vbh, N_, 16 + lm, j0, hf);
    v16h bv2 = ld_tile(Vbh, N_, 32 + lm, j0, hf);
    v16h bv3 = ld_tile(Vbh, N_, 48 + lm, j0, hf);
    if (j0 + 32 < N_) {
      const int jn = j0 + 32;
      k00 = ld_tile(Kbh, D_, jn +      lm,  0, hf);
      k01 = ld_tile(Kbh, D_, jn +      lm, 32, hf);
      k10 = ld_tile(Kbh, D_, jn + 16 + lm,  0, hf);
      k11 = ld_tile(Kbh, D_, jn + 16 + lm, 32, hf);
    }

    // Online softmax (rows live across the 16-lane halves).
    float p0[8], p1[8];
    for (int v = 0; v < 8; ++v) {
      float a  = s0[v] * scale;
      float bb = s1[v] * scale;
      float mx = fmaxf(a, bb);
      mx = fmaxf(mx, __shfl_xor(mx, 1, 32));
      mx = fmaxf(mx, __shfl_xor(mx, 2, 32));
      mx = fmaxf(mx, __shfl_xor(mx, 4, 32));
      mx = fmaxf(mx, __shfl_xor(mx, 8, 32));
      float mnew = fmaxf(mrow[v], mx);
      float corr = __expf(mrow[v] - mnew);
      float e0 = __expf(a  - mnew);
      float e1 = __expf(bb - mnew);
      float rs = e0 + e1;
      rs += __shfl_xor(rs, 1, 32);
      rs += __shfl_xor(rs, 2, 32);
      rs += __shfl_xor(rs, 4, 32);
      rs += __shfl_xor(rs, 8, 32);
      lrow[v] = lrow[v] * corr + rs;
      mrow[v] = mnew;
      p0[v] = e0; p1[v] = e1;
      o0[v] *= corr; o1[v] *= corr; o2[v] *= corr; o3[v] *= corr;
    }

    // C-layout -> A-layout transpose of P through per-wave LDS (in-order DS).
    for (int v = 0; v < 8; ++v) {
      int r = v + 8 * hf;
      lds[wv][r * 32 + lm]      = (h16)p0[v];
      lds[wv][r * 32 + 16 + lm] = (h16)p1[v];
    }
    __builtin_amdgcn_wave_barrier();
    v8h plo = *reinterpret_cast<const v8h*>(&lds[wv][lm * 32 + hf * 8]);
    v8h phi = *reinterpret_cast<const v8h*>(&lds[wv][lm * 32 + 16 + hf * 8]);
    v16h ap = __builtin_shufflevector(plo, phi, 0,1,2,3,4,5,6,7,8,9,10,11,12,13,14,15);
    __builtin_amdgcn_wave_barrier();

    // O += P V (V stored transposed so B columns are contiguous).
    o0 = wmma16(ap, bv0, o0);
    o1 = wmma16(ap, bv1, o1);
    o2 = wmma16(ap, bv2, o2);
    o3 = wmma16(ap, bv3, o3);
  }

  for (int v = 0; v < 8; ++v) {
    float inv = 1.0f / lrow[v];
    int n = n0 + v + 8 * hf;
    size_t base = ((size_t)b * N_ + n) * E_ + h * 64;
    CTX[base +  0 + lm] = (h16)(o0[v] * inv);
    CTX[base + 16 + lm] = (h16)(o1[v] * inv);
    CTX[base + 32 + lm] = (h16)(o2[v] * inv);
    CTX[base + 48 + lm] = (h16)(o3[v] * inv);
  }
}

// ------------------------------------------------------ output projection --
__global__ void __launch_bounds__(256) k_oproj(
    const h16* __restrict__ CTX, const h16* __restrict__ Wo,
    const float* __restrict__ bo, float* __restrict__ out)
{
  __shared__ h16 As[16][APAD];
  const int tid  = threadIdx.x;
  const int lane = tid & 31;
  const int wv   = tid >> 5;
  const int hf   = lane >> 4, lm = lane & 15;
  const int rt   = blockIdx.x;              // 0..511
  const int idx  = blockIdx.y * 8 + wv;     // 0..15
  const int f0   = idx * 64;

  stage_panel(As, CTX + (size_t)rt * 16 * E_, tid);
  const h16* arow = &As[lm][0];

  v8f c0 = {}, c1 = {}, c2 = {}, c3 = {};
  v16h a_c  = lds_a(arow, 0, hf);
  v16h b0_c = ld_tile(Wo, E_, f0 +  0 + lm, 0, hf);
  v16h b1_c = ld_tile(Wo, E_, f0 + 16 + lm, 0, hf);
  v16h b2_c = ld_tile(Wo, E_, f0 + 32 + lm, 0, hf);
  v16h b3_c = ld_tile(Wo, E_, f0 + 48 + lm, 0, hf);
  for (int k0 = 0; k0 < E_ - 32; k0 += 32) {
    const int kn = k0 + 32;
    v16h a_n  = lds_a(arow, kn, hf);
    v16h b0_n = ld_tile(Wo, E_, f0 +  0 + lm, kn, hf);
    v16h b1_n = ld_tile(Wo, E_, f0 + 16 + lm, kn, hf);
    v16h b2_n = ld_tile(Wo, E_, f0 + 32 + lm, kn, hf);
    v16h b3_n = ld_tile(Wo, E_, f0 + 48 + lm, kn, hf);
    c0 = wmma16(a_c, b0_c, c0);
    c1 = wmma16(a_c, b1_c, c1);
    c2 = wmma16(a_c, b2_c, c2);
    c3 = wmma16(a_c, b3_c, c3);
    a_c = a_n; b0_c = b0_n; b1_c = b1_n; b2_c = b2_n; b3_c = b3_n;
  }
  c0 = wmma16(a_c, b0_c, c0);
  c1 = wmma16(a_c, b1_c, c1);
  c2 = wmma16(a_c, b2_c, c2);
  c3 = wmma16(a_c, b3_c, c3);

  float bias0 = bo[f0 +  0 + lm];
  float bias1 = bo[f0 + 16 + lm];
  float bias2 = bo[f0 + 32 + lm];
  float bias3 = bo[f0 + 48 + lm];
  for (int v = 0; v < 8; ++v) {
    size_t grow = (size_t)(rt * 16 + v + 8 * hf);
    float* row = out + grow * E_;
    row[f0 +  0 + lm] = c0[v] + bias0;
    row[f0 + 16 + lm] = c1[v] + bias1;
    row[f0 + 32 + lm] = c2[v] + bias2;
    row[f0 + 48 + lm] = c3[v] + bias3;
  }
}

// ---------------------------------------------------------------- launch ---
extern "C" void kernel_launch(void* const* d_in, const int* in_sizes, int n_in,
                              void* d_out, int out_size, void* d_ws, size_t ws_size,
                              hipStream_t stream) {
  const float* x  = (const float*)d_in[0];
  const float* Wq = (const float*)d_in[1];
  const float* Wk = (const float*)d_in[2];
  const float* Wv = (const float*)d_in[3];
  const float* Wo = (const float*)d_in[4];
  const float* bo = (const float*)d_in[5];

  uint8_t* ws = (uint8_t*)d_ws;
  size_t off = 0;
  const size_t xBytes = (size_t)BN_ * E_ * sizeof(h16);          // 16 MB
  const size_t wBytes = (size_t)E_ * E_ * sizeof(h16);           // 2 MB
  const size_t tBytes = (size_t)B_ * H_ * N_ * D_ * sizeof(h16); // 16 MB
  h16* xh  = (h16*)(ws + off); off += xBytes;
  h16* Wqh = (h16*)(ws + off); off += wBytes;
  h16* Wkh = (h16*)(ws + off); off += wBytes;
  h16* Wvh = (h16*)(ws + off); off += wBytes;
  h16* Woh = (h16*)(ws + off); off += wBytes;
  h16* Qh  = (h16*)(ws + off); off += tBytes;
  h16* Kh  = (h16*)(ws + off); off += tBytes;
  h16* Vt  = (h16*)(ws + off); off += tBytes;
  h16* CTX = (h16*)(ws + off); off += tBytes;   // [B,N,E] f16

  const int nX = BN_ * E_;
  const int nW = E_ * E_;
  k_f32_to_f16<<<(nX / 4 + 255) / 256, 256, 0, stream>>>(x,  xh,  nX);
  k_f32_to_f16<<<(nW / 4 + 255) / 256, 256, 0, stream>>>(Wq, Wqh, nW);
  k_f32_to_f16<<<(nW / 4 + 255) / 256, 256, 0, stream>>>(Wk, Wkh, nW);
  k_f32_to_f16<<<(nW / 4 + 255) / 256, 256, 0, stream>>>(Wv, Wvh, nW);
  k_f32_to_f16<<<(nW / 4 + 255) / 256, 256, 0, stream>>>(Wo, Woh, nW);

  k_qkv_rope<<<dim3(BN_ / 16, 6), 256, 0, stream>>>(xh, Wqh, Wkh, Wvh, Qh, Kh, Vt);
  k_attn<<<dim3((B_ * H_ * (N_ / 16)) / 8), 256, 0, stream>>>(Qh, Kh, Vt, CTX);
  k_oproj<<<dim3(BN_ / 16, 2), 256, 0, stream>>>(CTX, Woh, bo, (float*)d_out);
}
